// AttentionBlock3D_37263136260902
// MI455X (gfx1250) — matrix operand order repaired and checked by reading, not compiled, not run on hardware
//
#include <hip/hip_runtime.h>
#include <hip/hip_bf16.h>
#include <stdint.h>

// ---------------------------------------------------------------------------
// AttentionBlock3D for MI455X (gfx1250, wave32, WMMA).
// b=2, c=256, n=4096 (16^3), 8 groups.
// Compute-bound (~40 GFLOP); flash-attention + v_wmma_f32_16x16x32_bf16.
// K/V tiles move via global_load_async_to_lds_b128 (ASYNCcnt), with the K
// tile double-buffered so the DMA overlaps the WMMA pipeline.
// ---------------------------------------------------------------------------

typedef __bf16 bf16;
typedef __attribute__((ext_vector_type(16))) __bf16 v16bf;
typedef __attribute__((ext_vector_type(8)))  __bf16 v8bf;
typedef __attribute__((ext_vector_type(8)))  float  v8f;

#define BATCH 2
#define CH    256
#define NVOX  4096
#define BC    32          // flash attention column tile
#define NT    (NVOX / BC) // 128 column tiles
#define LOG2E 1.44269504088896f

static __device__ __forceinline__ v8f wmma_bf16(v16bf a, v16bf b, v8f c) {
  // D = A(16x32) x B(32x16) + C, fp32 accum
  return __builtin_amdgcn_wmma_f32_16x16x32_bf16(
      /*neg_a=*/false, a, /*neg_b=*/false, b,
      /*c_mod=*/(short)0, c, /*reuse_a=*/false, /*reuse_b=*/false);
}

// A fragment (16x32 bf16): lane L holds row m=L%16; with half=L/16 the 16
// elements are K = half*16 + {0..15} (matches load_b_frag).
static __device__ __forceinline__ v16bf load_a_frag(const bf16* rowbase, int ldk, int kk) {
  const int lane = threadIdx.x & 31;
  const int m = lane & 15, half = lane >> 4;
  return *(const v16bf*)(rowbase + (size_t)m * ldk + kk * 32 + half * 16);
}

// B fragment (32x16 bf16) read from a transposed-B ("Bt[N][K]") row-major
// array: lane L holds column N=L%16, elements K = half*16 + {0..15}  -> one
// contiguous 32-byte load per lane.
static __device__ __forceinline__ v16bf load_b_frag(const bf16* bt, int ldk, int nBase, int kk) {
  const int lane = threadIdx.x & 31;
  const int nn = lane & 15, half = lane >> 4;
  return *(const v16bf*)(bt + (size_t)(nBase + nn) * ldk + kk * 32 + half * 16);
}

// --- CDNA5 async memory->LDS (ASYNCcnt) -----------------------------------
// VDST = VGPR holding LDS byte address; VADDR = 64-bit global address.
// Generic LDS pointers carry the LDS offset in their low 32 bits (ISA 10.2).
static __device__ __forceinline__ void async_ld_b128(unsigned lds_addr, const void* gaddr) {
  asm volatile("global_load_async_to_lds_b128 %0, %1, off"
               :: "v"(lds_addr), "v"(gaddr)
               : "memory");
}
// Per-wave async loads complete in order, so waiting ASYNCcnt<=N ensures the
// oldest (issued-first) transfers have landed.
static __device__ __forceinline__ void wait_async4() {
#if __has_builtin(__builtin_amdgcn_s_wait_asynccnt)
  __builtin_amdgcn_s_wait_asynccnt(4);
#else
  asm volatile("s_wait_asynccnt 4" ::: "memory");
#endif
}
static __device__ __forceinline__ void wait_async0() {
#if __has_builtin(__builtin_amdgcn_s_wait_asynccnt)
  __builtin_amdgcn_s_wait_asynccnt(0);
#else
  asm volatile("s_wait_asynccnt 0" ::: "memory");
#endif
}

// ---------------------------------------------------------------------------
// 1) GroupNorm statistics: one block per (batch, group); 32 ch x 4096 vox.
// ---------------------------------------------------------------------------
__global__ void gn_stats_kernel(const float* __restrict__ x, float* __restrict__ stats) {
  const int bg = blockIdx.x;                       // 0..15
  const float* xp = x + (size_t)bg * 32 * NVOX;
  float s = 0.f, s2 = 0.f;
  for (int i = threadIdx.x; i < 32 * NVOX; i += blockDim.x) {
    float v = xp[i];
    s += v; s2 += v * v;
  }
#pragma unroll
  for (int m = 16; m >= 1; m >>= 1) {
    s  += __shfl_xor(s,  m, 32);
    s2 += __shfl_xor(s2, m, 32);
  }
  __shared__ float ls[8], ls2[8];
  const int wid = threadIdx.x >> 5;
  if ((threadIdx.x & 31) == 0) { ls[wid] = s; ls2[wid] = s2; }
  __syncthreads();
  if (threadIdx.x == 0) {
    float ts = 0.f, ts2 = 0.f;
#pragma unroll
    for (int i = 0; i < 8; ++i) { ts += ls[i]; ts2 += ls2[i]; }
    const float inv_n = 1.0f / (32.0f * NVOX);
    float mean = ts * inv_n;
    float var  = ts2 * inv_n - mean * mean;
    stats[bg * 2 + 0] = mean;
    stats[bg * 2 + 1] = rsqrtf(var + 1e-5f);
  }
}

// ---------------------------------------------------------------------------
// 2) GroupNorm apply + transpose: hn_t[b][n][c] (bf16, WMMA-A friendly).
// ---------------------------------------------------------------------------
__global__ void gn_apply_kernel(const float* __restrict__ x,
                                const float* __restrict__ gamma,
                                const float* __restrict__ beta,
                                const float* __restrict__ stats,
                                bf16* __restrict__ hn) {
  int idx = blockIdx.x * blockDim.x + threadIdx.x;   // over 2*256*4096
  int pos = idx & (NVOX - 1);
  int ch  = (idx >> 12) & (CH - 1);
  int b   = idx >> 20;
  int bg  = b * 8 + (ch >> 5);
  float mean = stats[bg * 2 + 0], rstd = stats[bg * 2 + 1];
  float v = (x[idx] - mean) * rstd * gamma[ch] + beta[ch];
  hn[((size_t)b * NVOX + pos) * CH + ch] = (bf16)v;
}

// ---------------------------------------------------------------------------
// 3) fp32 -> bf16 weight conversion
// ---------------------------------------------------------------------------
__global__ void cvt_bf16_kernel(const float* __restrict__ w, bf16* __restrict__ o, int n) {
  int i = blockIdx.x * blockDim.x + threadIdx.x;
  if (i < n) o[i] = (bf16)w[i];
}

// ---------------------------------------------------------------------------
// Shared GEMM tile: 16 rows (one wave) x 256 cols, K=256.
// Preloads all 8 B fragments per 16-col tile so loads clause and the 8
// WMMAs issue back-to-back. MODE 0: out[n][c] bf16. MODE 1: out[c][n] bf16.
// ---------------------------------------------------------------------------
template <int MODE>
static __device__ __forceinline__ void proj_tile(
    const v16bf (&afr)[8], const bf16* __restrict__ W,
    const float* __restrict__ bias, bf16* __restrict__ out,
    int b, int mBase) {
  const int lane = threadIdx.x & 31;
  const int half = lane >> 4, nn = lane & 15;
  for (int nt = 0; nt < 16; ++nt) {
    v16bf bfr[8];
#pragma unroll
    for (int kk = 0; kk < 8; ++kk) bfr[kk] = load_b_frag(W, CH, nt * 16, kk);
    const float bv = bias[nt * 16 + nn];
    v8f acc;
#pragma unroll
    for (int r = 0; r < 8; ++r) acc[r] = bv;
#pragma unroll
    for (int kk = 0; kk < 8; ++kk) acc = wmma_bf16(afr[kk], bfr[kk], acc);
#pragma unroll
    for (int r = 0; r < 8; ++r) {
      const int row = mBase + r + half * 8;
      const int col = nt * 16 + nn;
      if (MODE == 0) out[((size_t)b * NVOX + row) * CH + col] = (bf16)acc[r];
      else           out[((size_t)b * CH + col) * NVOX + row] = (bf16)acc[r];
    }
  }
}

// ---------------------------------------------------------------------------
// 4) QKV projection. Block = 128 threads (4 waves), wave = 16 output rows.
//    Q,K -> [b][n][c] bf16 ; V -> transposed [b][c][n] bf16 (for P*V).
// ---------------------------------------------------------------------------
__global__ __launch_bounds__(128) void qkv_proj_kernel(
    const bf16* __restrict__ hn,
    const bf16* __restrict__ wq, const float* __restrict__ bq,
    const bf16* __restrict__ wk, const float* __restrict__ bk,
    const bf16* __restrict__ wv, const float* __restrict__ bv,
    bf16* __restrict__ Q, bf16* __restrict__ K, bf16* __restrict__ VT) {
  const int wave = threadIdx.x >> 5;
  const int b = blockIdx.x >> 6;                     // 64 blocks per batch
  const int mBase = (blockIdx.x & 63) * 64 + wave * 16;

  const bf16* arow = hn + ((size_t)b * NVOX + mBase) * CH;
  v16bf afr[8];
#pragma unroll
  for (int kk = 0; kk < 8; ++kk) afr[kk] = load_a_frag(arow, CH, kk);

  proj_tile<0>(afr, wq, bq, Q,  b, mBase);
  proj_tile<0>(afr, wk, bk, K,  b, mBase);
  proj_tile<1>(afr, wv, bv, VT, b, mBase);
}

// ---------------------------------------------------------------------------
// 5) Flash attention. Block = 256 threads (8 waves), 128 query rows / block.
//    LDS: K ping-pong 2x[32][256] + Vt [256][32] + P [8][16][32] = 56 KB.
//    Async copy pipeline per iteration (per-wave, in-order completion):
//      issue V(jt) [4 ops]  -> single V buffer
//      issue K(jt+1) [4 ops]-> alternate K buffer    (overlaps all compute)
//      S = Q*K^T, softmax, stage P
//      s_wait_asynccnt 4    (V landed)  + barrier
//      O += P*V
//      s_wait_asynccnt 0    (K(jt+1) landed) + barrier
// ---------------------------------------------------------------------------
__global__ __launch_bounds__(256) void flash_attn_kernel(
    const bf16* __restrict__ Q, const bf16* __restrict__ K,
    const bf16* __restrict__ VT, bf16* __restrict__ O) {
  __shared__ __align__(64) bf16 sK[2][BC * CH];     // ping-pong [j][c]
  __shared__ __align__(64) bf16 sV[CH * BC];        // [c][j]
  __shared__ __align__(64) bf16 sP[8 * 16 * BC];    // per wave 16x32

  const int wave = threadIdx.x >> 5;
  const int lane = threadIdx.x & 31;
  const int half = lane >> 4, nn = lane & 15;
  const int b = blockIdx.x >> 5;                     // 32 row-blocks / batch
  const int mBase = (blockIdx.x & 31) * 128 + wave * 16;

  const bf16* qrow = Q + ((size_t)b * NVOX + mBase) * CH;
  v16bf qfr[8];
#pragma unroll
  for (int kk = 0; kk < 8; ++kk) qfr[kk] = load_a_frag(qrow, CH, kk);

  float mrow[8], lrow[8];
  v8f oacc[16];
#pragma unroll
  for (int r = 0; r < 8; ++r) { mrow[r] = -3.0e38f; lrow[r] = 0.f; }
#pragma unroll
  for (int ct = 0; ct < 16; ++ct) {
    v8f z;
#pragma unroll
    for (int r = 0; r < 8; ++r) z[r] = 0.f;
    oacc[ct] = z;
  }

  const unsigned lKbase = (unsigned)(uintptr_t)(void*)&sK[0][0];
  const unsigned lVbase = (unsigned)(uintptr_t)(void*)sV;
  const float sc = 0.0625f;                          // 1/sqrt(256)

  // async issue helpers (4 x b128 per thread each)
  auto issueK = [&](int jt, int buf) {
    const bf16* gK = K + ((size_t)b * NVOX + jt * BC) * CH;  // contiguous 16KB
    const unsigned lbase = lKbase + (unsigned)buf * (BC * CH * 2);
#pragma unroll
    for (int i = 0; i < 4; ++i) {
      const int chunk = threadIdx.x + 256 * i;               // 1024 x 16B
      async_ld_b128(lbase + chunk * 16, gK + chunk * 8);
    }
  };
  auto issueV = [&](int jt) {
    const bf16* gV = VT + ((size_t)b * CH + threadIdx.x) * NVOX + jt * BC;
    const unsigned l = lVbase + threadIdx.x * (BC * 2);
#pragma unroll
    for (int i = 0; i < 4; ++i)
      async_ld_b128(l + i * 16, gV + i * 8);
  };

  // prologue: land K tile 0
  issueK(0, 0);
  wait_async0();
  __syncthreads();

  for (int jt = 0; jt < NT; ++jt) {
    const bf16* curK = sK[jt & 1];

    // in-flight DMA: V for this tile, K for the next (wraps at the end so
    // the ASYNCcnt discipline stays uniform; the extra tile-0 load is benign)
    issueV(jt);
    issueK((jt + 1) & (NT - 1), (jt + 1) & 1);

    // --- S = Q * K^T  (16 x 32 per wave) ---
    v8f s0, s1;
#pragma unroll
    for (int r = 0; r < 8; ++r) { s0[r] = 0.f; s1[r] = 0.f; }
#pragma unroll
    for (int kk = 0; kk < 8; ++kk) {
      s0 = wmma_bf16(qfr[kk], load_b_frag(curK, CH, 0,  kk), s0);
      s1 = wmma_bf16(qfr[kk], load_b_frag(curK, CH, 16, kk), s1);
    }

    // --- online softmax; row m = r + half*8 lives across 16 lanes ---
    float alpha[8];
    bf16* pbase = sP + wave * 16 * BC;
#pragma unroll
    for (int r = 0; r < 8; ++r) {
      float v0 = s0[r] * sc, v1 = s1[r] * sc;
      float mx = fmaxf(v0, v1);
#pragma unroll
      for (int msk = 1; msk < 16; msk <<= 1) mx = fmaxf(mx, __shfl_xor(mx, msk, 32));
      float mnew = fmaxf(mrow[r], mx);
      float a  = __builtin_amdgcn_exp2f((mrow[r] - mnew) * LOG2E);
      float p0 = __builtin_amdgcn_exp2f((v0 - mnew) * LOG2E);
      float p1 = __builtin_amdgcn_exp2f((v1 - mnew) * LOG2E);
      float ps = p0 + p1;
#pragma unroll
      for (int msk = 1; msk < 16; msk <<= 1) ps += __shfl_xor(ps, msk, 32);
      alpha[r] = a;
      lrow[r] = lrow[r] * a + ps;
      mrow[r] = mnew;
      const int row = r + half * 8;
      pbase[row * BC + nn]      = (bf16)p0;
      pbase[row * BC + 16 + nn] = (bf16)p1;
    }

    // V issued first -> ASYNCcnt<=4 means V landed (K(next) may be in flight);
    // barrier turns the per-wave guarantee into a block-wide one + makes P
    // visible.
    wait_async4();
    __syncthreads();

    // --- O = diag(alpha)*O + P * V ---
    v16bf pfr = load_a_frag(pbase, BC, 0);
#pragma unroll
    for (int ct = 0; ct < 16; ++ct) {
      v8f acc = oacc[ct];
#pragma unroll
      for (int r = 0; r < 8; ++r) acc[r] *= alpha[r];
      oacc[ct] = wmma_bf16(pfr, load_b_frag(sV, BC, ct * 16, 0), acc);
    }

    // K(next) landed everywhere + all waves done reading this iteration's LDS
    wait_async0();
    __syncthreads();
  }

  // --- epilogue: O[b][n][c] bf16 ---
#pragma unroll
  for (int ct = 0; ct < 16; ++ct) {
#pragma unroll
    for (int r = 0; r < 8; ++r) {
      const int row = mBase + r + half * 8;
      const int col = ct * 16 + nn;
      O[((size_t)b * NVOX + row) * CH + col] = (bf16)(oacc[ct][r] / lrow[r]);
    }
  }
}

// ---------------------------------------------------------------------------
// 6) Output projection + residual: y[b][c][n] = x + Wp * O + bp
// ---------------------------------------------------------------------------
__global__ __launch_bounds__(128) void out_proj_kernel(
    const bf16* __restrict__ O, const bf16* __restrict__ wp,
    const float* __restrict__ bp, const float* __restrict__ x,
    float* __restrict__ y) {
  const int wave = threadIdx.x >> 5;
  const int lane = threadIdx.x & 31;
  const int half = lane >> 4, nn = lane & 15;
  const int b = blockIdx.x >> 6;
  const int mBase = (blockIdx.x & 63) * 64 + wave * 16;

  const bf16* arow = O + ((size_t)b * NVOX + mBase) * CH;
  v16bf afr[8];
#pragma unroll
  for (int kk = 0; kk < 8; ++kk) afr[kk] = load_a_frag(arow, CH, kk);

  for (int nt = 0; nt < 16; ++nt) {
    v16bf bfr[8];
#pragma unroll
    for (int kk = 0; kk < 8; ++kk) bfr[kk] = load_b_frag(wp, CH, nt * 16, kk);
    const float bias = bp[nt * 16 + nn];
    v8f acc;
#pragma unroll
    for (int r = 0; r < 8; ++r) acc[r] = bias;
#pragma unroll
    for (int kk = 0; kk < 8; ++kk) acc = wmma_bf16(afr[kk], bfr[kk], acc);
#pragma unroll
    for (int r = 0; r < 8; ++r) {
      const int row = mBase + r + half * 8;
      const int chn = nt * 16 + nn;
      const size_t idx = ((size_t)b * CH + chn) * NVOX + row;
      y[idx] = x[idx] + acc[r];
    }
  }
}

// ---------------------------------------------------------------------------
// host-side launch
// ---------------------------------------------------------------------------
extern "C" void kernel_launch(void* const* d_in, const int* in_sizes, int n_in,
                              void* d_out, int out_size, void* d_ws, size_t ws_size,
                              hipStream_t stream) {
  (void)in_sizes; (void)n_in; (void)out_size; (void)ws_size;

  const float* x     = (const float*)d_in[0];
  const float* gamma = (const float*)d_in[1];
  const float* beta  = (const float*)d_in[2];
  const float* Wq    = (const float*)d_in[3];
  const float* bq    = (const float*)d_in[4];
  const float* Wk    = (const float*)d_in[5];
  const float* bk    = (const float*)d_in[6];
  const float* Wv    = (const float*)d_in[7];
  const float* bv    = (const float*)d_in[8];
  const float* Wp    = (const float*)d_in[9];
  const float* bp    = (const float*)d_in[10];
  float* y = (float*)d_out;

  // workspace layout (bytes; each 256-aligned)
  constexpr size_t SZ_BNC = (size_t)BATCH * NVOX * CH * sizeof(bf16);  // 4 MiB
  constexpr size_t SZ_W   = (size_t)CH * CH * sizeof(bf16);            // 128 KiB
  constexpr size_t OFF_STATS = 0;
  constexpr size_t OFF_HN = 256;
  constexpr size_t OFF_WQ = OFF_HN + SZ_BNC;
  constexpr size_t OFF_WK = OFF_WQ + SZ_W;
  constexpr size_t OFF_WV = OFF_WK + SZ_W;
  constexpr size_t OFF_WP = OFF_WV + SZ_W;
  constexpr size_t OFF_Q  = OFF_WP + SZ_W;
  constexpr size_t OFF_K  = OFF_Q + SZ_BNC;
  constexpr size_t OFF_VT = OFF_K + SZ_BNC;
  constexpr size_t OFF_O  = OFF_VT + SZ_BNC;

  char* ws = (char*)d_ws;
  float* stats = (float*)(ws + OFF_STATS);
  bf16*  hn    = (bf16*)(ws + OFF_HN);
  bf16*  wq_b  = (bf16*)(ws + OFF_WQ);
  bf16*  wk_b  = (bf16*)(ws + OFF_WK);
  bf16*  wv_b  = (bf16*)(ws + OFF_WV);
  bf16*  wp_b  = (bf16*)(ws + OFF_WP);
  bf16*  Qb    = (bf16*)(ws + OFF_Q);
  bf16*  Kb    = (bf16*)(ws + OFF_K);
  bf16*  VTb   = (bf16*)(ws + OFF_VT);
  bf16*  Ob    = (bf16*)(ws + OFF_O);

  gn_stats_kernel<<<BATCH * 8, 256, 0, stream>>>(x, stats);
  gn_apply_kernel<<<(BATCH * CH * NVOX) / 256, 256, 0, stream>>>(x, gamma, beta, stats, hn);

  cvt_bf16_kernel<<<(CH * CH) / 256, 256, 0, stream>>>(Wq, wq_b, CH * CH);
  cvt_bf16_kernel<<<(CH * CH) / 256, 256, 0, stream>>>(Wk, wk_b, CH * CH);
  cvt_bf16_kernel<<<(CH * CH) / 256, 256, 0, stream>>>(Wv, wv_b, CH * CH);
  cvt_bf16_kernel<<<(CH * CH) / 256, 256, 0, stream>>>(Wp, wp_b, CH * CH);

  // 64 rows per block, 4 waves
  qkv_proj_kernel<<<BATCH * (NVOX / 64), 128, 0, stream>>>(
      hn, wq_b, bq, wk_b, bk, wv_b, bv, Qb, Kb, VTb);

  // 128 query rows per block, 8 waves
  flash_attn_kernel<<<BATCH * (NVOX / 128), 256, 0, stream>>>(Qb, Kb, VTb, Ob);

  out_proj_kernel<<<BATCH * (NVOX / 64), 128, 0, stream>>>(Ob, wp_b, bp, x, y);
}
